// BiLSTMDecoderModel_89051851915328
// MI455X (gfx1250) — compile-verified, matrix operand order checked by reading, not run on hardware
//
#include <hip/hip_runtime.h>
#include <hip/hip_bf16.h>
#include <math.h>

// ---------------- model dims ----------------
#define B_   32
#define S_   128
#define W_   16
#define E_   256
#define CD_  64
#define H_   512
#define P_   256
#define NC_  6
#define CK_  8
#define WK_  64
#define XC_  (E_ + CD_)      // 320
#define KW_  (S_ * XC_)      // 40960  (collapsed word-conv K)
#define KC_  (W_ * CD_)      // 1024   (collapsed char-conv K)

typedef __bf16 bf16t;
typedef __attribute__((ext_vector_type(16))) __bf16        v16bf;
typedef __attribute__((ext_vector_type(8)))  float         v8f;
typedef __attribute__((ext_vector_type(4)))  unsigned int  u32x4;

union Frag { v16bf v; u32x4 q[2]; };

#define GID ((int)(blockIdx.x * blockDim.x + threadIdx.x))

__device__ __forceinline__ float sigf(float x) { return 1.0f / (1.0f + expf(-x)); }

// ---------------------------------------------------------------------------
// Generic NT WMMA GEMM: C[M,N] = act(scale * (A[M,K] * Bt[N,K]^T) + bias[N])
// Register-blocked: one wave computes a 32x32 block = 2x2 tiles of 16x16,
// so each k-step is 8x global_load_b128 feeding 4 independent
// v_wmma_f32_16x16x32_bf16 (shared A/B fragments, ~16 flops/byte).
// OOB rows handled by clamping (results discarded at store) -- no divergent
// zero-fill. Requires: K % 32 == 0, N % 32 == 0, rows 16B-aligned (true here).
// ---------------------------------------------------------------------------
template<int ACT, int OUTBF>
__global__ __launch_bounds__(256)
void gemm_nt(const bf16t* __restrict__ A, const bf16t* __restrict__ Bt,
             const float* __restrict__ bias, void* __restrict__ Cv,
             int M, int N, int K, int ldc, float scale) {
  const int lane = threadIdx.x & 31;
  const int wave = (int)(blockIdx.x * (blockDim.x >> 5)) + (threadIdx.x >> 5);
  const int ntN = (N + 31) >> 5;
  const int ntM = (M + 31) >> 5;
  if (wave >= ntM * ntN) return;                // wave-uniform exit: EXEC full at WMMA
  const int mt = wave / ntN, nt = wave - mt * ntN;
  const int m0 = mt << 5, n0 = nt << 5;
  const int khalf = (lane & 16) ? 8 : 0;        // 16-bit A/B fragment layout (ISA 7.12.2)
  int mA0 = m0 + (lane & 15);      if (mA0 >= M) mA0 = M - 1;   // clamp; discarded at store
  int mA1 = m0 + 16 + (lane & 15); if (mA1 >= M) mA1 = M - 1;
  const int nB0 = n0 + (lane & 15);
  const int nB1 = nB0 + 16;                     // N % 32 == 0 -> always in range
  const bf16t* arow0 = A  + (size_t)mA0 * (size_t)K + khalf;
  const bf16t* arow1 = A  + (size_t)mA1 * (size_t)K + khalf;
  const bf16t* brow0 = Bt + (size_t)nB0 * (size_t)K + khalf;
  const bf16t* brow1 = Bt + (size_t)nB1 * (size_t)K + khalf;
  v8f acc00 = {0.f,0.f,0.f,0.f,0.f,0.f,0.f,0.f};
  v8f acc01 = acc00, acc10 = acc00, acc11 = acc00;
  for (int k = 0; k < K; k += 32) {
    Frag a0, a1, b0, b1;
    a0.q[0] = *(const u32x4*)(arow0 + k);
    a0.q[1] = *(const u32x4*)(arow0 + k + 16);
    a1.q[0] = *(const u32x4*)(arow1 + k);
    a1.q[1] = *(const u32x4*)(arow1 + k + 16);
    b0.q[0] = *(const u32x4*)(brow0 + k);
    b0.q[1] = *(const u32x4*)(brow0 + k + 16);
    b1.q[0] = *(const u32x4*)(brow1 + k);
    b1.q[1] = *(const u32x4*)(brow1 + k + 16);
    acc00 = __builtin_amdgcn_wmma_f32_16x16x32_bf16(false, a0.v, false, b0.v,
                                                    (short)0, acc00, false, false);
    acc01 = __builtin_amdgcn_wmma_f32_16x16x32_bf16(false, a0.v, false, b1.v,
                                                    (short)0, acc01, false, false);
    acc10 = __builtin_amdgcn_wmma_f32_16x16x32_bf16(false, a1.v, false, b0.v,
                                                    (short)0, acc10, false, false);
    acc11 = __builtin_amdgcn_wmma_f32_16x16x32_bf16(false, a1.v, false, b1.v,
                                                    (short)0, acc11, false, false);
  }
  const int mside = (lane & 16) ? 8 : 0;        // f32 C/D layout (ISA 7.12.2)
#pragma unroll
  for (int half = 0; half < 2; ++half) {
    const v8f* accs0 = half ? &acc10 : &acc00;
    const v8f* accs1 = half ? &acc11 : &acc01;
    const int mbase = m0 + (half ? 16 : 0) + mside;
#pragma unroll
    for (int v = 0; v < 8; ++v) {
      const int mm = mbase + v;
      if (mm < M) {
        float r0 = (*accs0)[v] * scale;
        float r1 = (*accs1)[v] * scale;
        if (bias) { r0 += bias[nB0]; r1 += bias[nB1]; }
        if (ACT == 1) { r0 = tanhf(r0); r1 = tanhf(r1); }
        if (OUTBF) {
          ((bf16t*)Cv)[(size_t)mm * ldc + nB0] = (bf16t)r0;
          ((bf16t*)Cv)[(size_t)mm * ldc + nB1] = (bf16t)r1;
        } else {
          ((float*)Cv)[(size_t)mm * ldc + nB0] = r0;
          ((float*)Cv)[(size_t)mm * ldc + nB1] = r1;
        }
      }
    }
  }
}

// ---------------- elementwise / prep kernels ----------------

__global__ void k_f2bf(const float* __restrict__ s, bf16t* __restrict__ d, int n) {
  int i = GID; if (i < n) d[i] = (bf16t)s[i];
}

// x[:, 0:E] = tanh(embed_w[seq])   (bf16, row = b*S+s, ld = XC_)
__global__ void k_word_embed(const int* __restrict__ seq, const float* __restrict__ embed_w,
                             bf16t* __restrict__ x) {
  int i = GID; if (i >= B_ * S_ * E_) return;
  int e = i % E_, bs = i / E_;
  int tok = seq[bs];
  x[(size_t)bs * XC_ + e] = (bf16t)tanhf(embed_w[(size_t)tok * E_ + e]);
}

// ce[bs, wpos*CD+ic] = char_embed_w[seq_char[bs,wpos], ic]   (bf16)
__global__ void k_char_gather(const int* __restrict__ seq_char, const float* __restrict__ cemb,
                              bf16t* __restrict__ ce) {
  int i = GID; if (i >= B_ * S_ * W_ * CD_) return;
  int ic = i % CD_, r = i / CD_;
  int wpos = r % W_, bs = r / W_;
  int tok = seq_char[bs * W_ + wpos];
  ce[(size_t)bs * KC_ + wpos * CD_ + ic] = (bf16t)cemb[tok * CD_ + ic];
}

// char conv collapsed over output positions: wsum[oc, wpos*CD+ic] = sum_k w[oc,ic,k]
__global__ void k_char_wsum(const float* __restrict__ w, bf16t* __restrict__ wsum) {
  int i = GID; if (i >= CD_ * KC_) return;
  int t = i % KC_, oc = i / KC_;
  int wpos = t / CD_, ic = t % CD_;
  int k0 = wpos - (W_ - CK_); if (k0 < 0) k0 = 0;        // max(0, wpos-8)
  int k1 = wpos; if (k1 > CK_ - 1) k1 = CK_ - 1;         // min(7, wpos)
  float s = 0.f;
  for (int k = k0; k <= k1; ++k) s += w[((size_t)oc * CD_ + ic) * CK_ + k];
  wsum[(size_t)oc * KC_ + t] = (bf16t)s;
}

// word conv collapsed: wsum[oc, s*XC+c] = sum_{k=max(0,s-64)}^{min(63,s)} w[oc,c,k]
__global__ void k_word_wsum(const float* __restrict__ w, bf16t* __restrict__ wsum) {
  int i = GID; if (i >= E_ * KW_) return;
  int t = i % KW_, oc = i / KW_;
  int s = t / XC_, c = t % XC_;
  int k0 = s - (S_ - WK_); if (k0 < 0) k0 = 0;           // max(0, s-64)
  int k1 = s; if (k1 > WK_ - 1) k1 = WK_ - 1;            // min(63, s)
  float acc = 0.f;
  const float* wr = w + ((size_t)oc * XC_ + c) * WK_;
  for (int k = k0; k <= k1; ++k) acc += wr[k];
  wsum[(size_t)oc * KW_ + t] = (bf16t)acc;
}

// LSTM gate update for both directions at step t
__global__ void k_lstm_gate(int t,
                            const float* __restrict__ gi_f, const float* __restrict__ gi_b,
                            const float* __restrict__ gh_f, const float* __restrict__ gh_b,
                            float* __restrict__ c_f, float* __restrict__ c_b,
                            bf16t* __restrict__ h_f, bf16t* __restrict__ h_b,
                            bf16t* __restrict__ seq_repr) {
  int i = GID; if (i >= 2 * B_ * H_) return;
  int j = i % H_;
  int b = (i / H_) % B_;
  int dir = i / (H_ * B_);
  const float *gi, *gh; float* cbuf; bf16t* hbuf; int s_in, t_out, hoff;
  if (dir == 0) { s_in = t; t_out = t; gi = gi_f; gh = gh_f; cbuf = c_f; hbuf = h_f; hoff = 0; }
  else { s_in = (t == 0) ? 0 : (S_ - t); t_out = S_ - 1 - t;
         gi = gi_b; gh = gh_b; cbuf = c_b; hbuf = h_b; hoff = H_; }
  size_t gib = (size_t)(b * S_ + s_in) * (4 * H_);
  size_t ghb = (size_t)b * (4 * H_);
  float gI = gi[gib + j]          + gh[ghb + j];
  float gF = gi[gib + H_ + j]     + gh[ghb + H_ + j];
  float gG = gi[gib + 2 * H_ + j] + gh[ghb + 2 * H_ + j];
  float gO = gi[gib + 3 * H_ + j] + gh[ghb + 3 * H_ + j];
  float c = sigf(gF) * cbuf[b * H_ + j] + sigf(gI) * tanhf(gG);
  float h = sigf(gO) * tanhf(c);
  cbuf[b * H_ + j] = c;
  hbuf[b * H_ + j] = (bf16t)h;
  seq_repr[(size_t)(b * S_ + t_out) * (2 * H_) + hoff + j] = (bf16t)h;
}

// a_bf[c,h] = tanh( sum_e tanh(tanh(class_embed[c,e])) * attend[e,h] )
__global__ void k_class_attn(const float* __restrict__ cemb, const float* __restrict__ attend,
                             bf16t* __restrict__ a_bf) {
  int i = GID; if (i >= NC_ * 2 * H_) return;
  int h = i % (2 * H_), c = i / (2 * H_);
  float acc = 0.f;
  for (int e = 0; e < E_; ++e)
    acc += tanhf(tanhf(cemb[c * E_ + e])) * attend[(size_t)e * (2 * H_) + h];
  a_bf[(size_t)c * (2 * H_) + h] = (bf16t)tanhf(acc);
}

// att_ctx[c,b,h] = sum_t tanh(mask * scores[c, b*S+t]) * seq_repr[b,t,h]  -> inp[:,256:1280]
__global__ void k_att_ctx(const int* __restrict__ seq, const float* __restrict__ scores,
                          const bf16t* __restrict__ seq_repr, bf16t* __restrict__ inp) {
  int i = GID; if (i >= NC_ * B_ * 2 * H_) return;
  int h = i % (2 * H_);
  int b = (i / (2 * H_)) % B_;
  int c = i / (2 * H_ * B_);
  float acc = 0.f;
  for (int t = 0; t < S_; ++t) {
    if (seq[b * S_ + t] > 0) {
      float sc = tanhf(scores[(size_t)c * (B_ * S_) + b * S_ + t]);
      acc += sc * (float)seq_repr[(size_t)(b * S_ + t) * (2 * H_) + h];
    }
  }
  inp[(size_t)(c * B_ + b) * (E_ + 2 * H_) + E_ + h] = (bf16t)acc;
}

// inp[:,0:E] = conv_seq (broadcast over classes)
__global__ void k_fill_inp(const float* __restrict__ conv_seq, bf16t* __restrict__ inp) {
  int i = GID; if (i >= NC_ * B_ * E_) return;
  int e = i % E_;
  int b = (i / E_) % B_;
  int c = i / (E_ * B_);
  inp[(size_t)(c * B_ + b) * (E_ + 2 * H_) + e] = (bf16t)conv_seq[b * E_ + e];
}

// decoder h0 = seq_repr[:, S-1, :]
__global__ void k_init_hdec(const bf16t* __restrict__ seq_repr, bf16t* __restrict__ h) {
  int i = GID; if (i >= B_ * 2 * H_) return;
  int j = i % (2 * H_), b = i / (2 * H_);
  h[i] = seq_repr[(size_t)(b * S_ + (S_ - 1)) * (2 * H_) + j];
}

// GRU gate at class-step c (in-place on bf16 h)
__global__ void k_gru_gate(int c, const float* __restrict__ gi, const float* __restrict__ gh,
                           bf16t* __restrict__ h) {
  int i = GID; if (i >= B_ * 2 * H_) return;
  int j = i % (2 * H_), b = i / (2 * H_);
  size_t gib = (size_t)(c * B_ + b) * (6 * H_);
  size_t ghb = (size_t)b * (6 * H_);
  float r = sigf(gi[gib + j]              + gh[ghb + j]);
  float z = sigf(gi[gib + 2 * H_ + j]     + gh[ghb + 2 * H_ + j]);
  float n = tanhf(gi[gib + 4 * H_ + j] + r * gh[ghb + 4 * H_ + j]);
  float hold = (float)h[i];
  h[i] = (bf16t)tanhf((1.f - z) * n + z * hold);
}

// final classifier + log_softmax at class-step c
__global__ void k_cls_out(int c, const float* __restrict__ proj_out,
                          const float* __restrict__ cls_w, const float* __restrict__ cls_b,
                          float* __restrict__ out) {
  int b = GID; if (b >= B_) return;
  float l0 = cls_b[0], l1 = cls_b[1];
  for (int p = 0; p < P_; ++p) {
    float v = proj_out[b * P_ + p];
    l0 += v * cls_w[p];
    l1 += v * cls_w[P_ + p];
  }
  float m = fmaxf(l0, l1);
  float lse = m + logf(expf(l0 - m) + expf(l1 - m));
  out[(size_t)(c * B_ + b) * 2 + 0] = l0 - lse;
  out[(size_t)(c * B_ + b) * 2 + 1] = l1 - lse;
}

// ---------------- host side ----------------

static inline int cdiv(long long a, long long b) { return (int)((a + b - 1) / b); }

template<int ACT, int OUTBF>
static void gemm(const bf16t* A, const bf16t* Bt, const float* bias, void* C,
                 int M, int N, int K, int ldc, float scale, hipStream_t s) {
  int tiles = ((M + 31) / 32) * ((N + 31) / 32);   // 32x32 blocks per wave
  int blocks = cdiv(tiles, 8);                     // 8 waves (256 threads) per block
  gemm_nt<ACT, OUTBF><<<blocks, 256, 0, s>>>(A, Bt, bias, C, M, N, K, ldc, scale);
}

extern "C" void kernel_launch(void* const* d_in, const int* in_sizes, int n_in,
                              void* d_out, int out_size, void* d_ws, size_t ws_size,
                              hipStream_t stream) {
  (void)in_sizes; (void)n_in; (void)out_size; (void)ws_size;
  const int*   seq       = (const int*)  d_in[0];
  const int*   seq_char  = (const int*)  d_in[1];
  const float* embed_w   = (const float*)d_in[3];
  const float* cemb_w    = (const float*)d_in[4];
  const float* clsemb_w  = (const float*)d_in[5];
  const float* cconv_w   = (const float*)d_in[6];
  const float* cconv_b   = (const float*)d_in[7];
  const float* wconv_w   = (const float*)d_in[8];
  const float* wconv_b   = (const float*)d_in[9];
  const float* f_Wih     = (const float*)d_in[10];
  const float* f_Whh     = (const float*)d_in[11];
  const float* f_b       = (const float*)d_in[12];
  const float* b_Wih     = (const float*)d_in[13];
  const float* b_Whh     = (const float*)d_in[14];
  const float* b_b       = (const float*)d_in[15];
  const float* attend    = (const float*)d_in[16];
  const float* d_Wih     = (const float*)d_in[17];
  const float* d_Whh     = (const float*)d_in[18];
  const float* d_bih     = (const float*)d_in[19];
  const float* d_bhh     = (const float*)d_in[20];
  const float* proj_w    = (const float*)d_in[21];
  const float* proj_b    = (const float*)d_in[22];
  const float* cls_w     = (const float*)d_in[23];
  const float* cls_b     = (const float*)d_in[24];
  float* out = (float*)d_out;

  // bump allocator on d_ws (256B-aligned blocks -> all GEMM operands 16B-aligned)
  char* ws = (char*)d_ws;
  size_t off = 0;
  auto alloc = [&](size_t bytes) -> void* {
    off = (off + 255) & ~(size_t)255;
    void* p = ws + off;
    off += bytes;
    return p;
  };
  const int BS = B_ * S_;                       // 4096
  bf16t* x_bf      = (bf16t*)alloc((size_t)BS * XC_ * 2);
  bf16t* ce_bf     = (bf16t*)alloc((size_t)BS * KC_ * 2);
  bf16t* wsum_c    = (bf16t*)alloc((size_t)CD_ * KC_ * 2);
  bf16t* wsum_w    = (bf16t*)alloc((size_t)E_ * KW_ * 2);
  float* conv_seq  = (float*)alloc((size_t)B_ * E_ * 4);
  bf16t* fWih_bf   = (bf16t*)alloc((size_t)4 * H_ * XC_ * 2);
  bf16t* bWih_bf   = (bf16t*)alloc((size_t)4 * H_ * XC_ * 2);
  bf16t* fWhh_bf   = (bf16t*)alloc((size_t)4 * H_ * H_ * 2);
  bf16t* bWhh_bf   = (bf16t*)alloc((size_t)4 * H_ * H_ * 2);
  float* gi_f      = (float*)alloc((size_t)BS * 4 * H_ * 4);
  float* gi_b      = (float*)alloc((size_t)BS * 4 * H_ * 4);
  float* gh_f      = (float*)alloc((size_t)B_ * 4 * H_ * 4);
  float* gh_b      = (float*)alloc((size_t)B_ * 4 * H_ * 4);
  bf16t* h_f       = (bf16t*)alloc((size_t)B_ * H_ * 2);
  bf16t* h_b       = (bf16t*)alloc((size_t)B_ * H_ * 2);
  float* c_f       = (float*)alloc((size_t)B_ * H_ * 4);
  float* c_b       = (float*)alloc((size_t)B_ * H_ * 4);
  bf16t* seq_repr  = (bf16t*)alloc((size_t)BS * 2 * H_ * 2);
  bf16t* a_bf      = (bf16t*)alloc((size_t)NC_ * 2 * H_ * 2);
  float* scores    = (float*)alloc((size_t)NC_ * BS * 4);
  bf16t* inp_bf    = (bf16t*)alloc((size_t)NC_ * B_ * (E_ + 2 * H_) * 2);
  bf16t* dWih_bf   = (bf16t*)alloc((size_t)6 * H_ * (E_ + 2 * H_) * 2);
  bf16t* dWhh_bf   = (bf16t*)alloc((size_t)6 * H_ * 2 * H_ * 2);
  float* gi_dec    = (float*)alloc((size_t)NC_ * B_ * 6 * H_ * 4);
  float* gh_dec    = (float*)alloc((size_t)B_ * 6 * H_ * 4);
  bf16t* h_dec     = (bf16t*)alloc((size_t)B_ * 2 * H_ * 2);
  bf16t* projw_bf  = (bf16t*)alloc((size_t)P_ * 2 * H_ * 2);
  float* proj_out  = (float*)alloc((size_t)B_ * P_ * 4);

  const int T = 256;  // elementwise block size

  // ---- weight conversions to bf16 ----
  k_f2bf<<<cdiv(4 * H_ * XC_, T), T, 0, stream>>>(f_Wih, fWih_bf, 4 * H_ * XC_);
  k_f2bf<<<cdiv(4 * H_ * XC_, T), T, 0, stream>>>(b_Wih, bWih_bf, 4 * H_ * XC_);
  k_f2bf<<<cdiv(4 * H_ * H_, T), T, 0, stream>>>(f_Whh, fWhh_bf, 4 * H_ * H_);
  k_f2bf<<<cdiv(4 * H_ * H_, T), T, 0, stream>>>(b_Whh, bWhh_bf, 4 * H_ * H_);
  k_f2bf<<<cdiv(6 * H_ * (E_ + 2 * H_), T), T, 0, stream>>>(d_Wih, dWih_bf, 6 * H_ * (E_ + 2 * H_));
  k_f2bf<<<cdiv(6 * H_ * 2 * H_, T), T, 0, stream>>>(d_Whh, dWhh_bf, 6 * H_ * 2 * H_);
  k_f2bf<<<cdiv(P_ * 2 * H_, T), T, 0, stream>>>(proj_w, projw_bf, P_ * 2 * H_);

  // ---- embeddings ----
  k_word_embed<<<cdiv(BS * E_, T), T, 0, stream>>>(seq, embed_w, x_bf);
  k_char_gather<<<cdiv(BS * W_ * CD_, T), T, 0, stream>>>(seq_char, cemb_w, ce_bf);

  // ---- char CNN (conv+mean collapsed into one GEMM), tanh, write into x[:,256:320] ----
  k_char_wsum<<<cdiv(CD_ * KC_, T), T, 0, stream>>>(cconv_w, wsum_c);
  gemm<1, 1>(ce_bf, wsum_c, cconv_b, x_bf + E_, BS, CD_, KC_, XC_, 1.f / 9.f, stream);

  // ---- word CNN (collapsed), tanh -> conv_seq [B,E] ----
  k_word_wsum<<<cdiv(E_ * KW_, T), T, 0, stream>>>(wconv_w, wsum_w);
  gemm<1, 0>(x_bf, wsum_w, wconv_b, conv_seq, B_, E_, KW_, E_, 1.f / 65.f, stream);

  // ---- LSTM input-side GEMMs (time-independent, all steps at once) ----
  gemm<0, 0>(x_bf, fWih_bf, f_b, gi_f, BS, 4 * H_, XC_, 4 * H_, 1.f, stream);
  gemm<0, 0>(x_bf, bWih_bf, b_b, gi_b, BS, 4 * H_, XC_, 4 * H_, 1.f, stream);

  // ---- BiLSTM recurrence ----
  hipMemsetAsync(h_f, 0, (size_t)B_ * H_ * 2, stream);
  hipMemsetAsync(h_b, 0, (size_t)B_ * H_ * 2, stream);
  hipMemsetAsync(c_f, 0, (size_t)B_ * H_ * 4, stream);
  hipMemsetAsync(c_b, 0, (size_t)B_ * H_ * 4, stream);
  for (int t = 0; t < S_; ++t) {
    gemm<0, 0>(h_f, fWhh_bf, nullptr, gh_f, B_, 4 * H_, H_, 4 * H_, 1.f, stream);
    gemm<0, 0>(h_b, bWhh_bf, nullptr, gh_b, B_, 4 * H_, H_, 4 * H_, 1.f, stream);
    k_lstm_gate<<<cdiv(2 * B_ * H_, T), T, 0, stream>>>(t, gi_f, gi_b, gh_f, gh_b,
                                                        c_f, c_b, h_f, h_b, seq_repr);
  }

  // ---- class attention ----
  k_class_attn<<<cdiv(NC_ * 2 * H_, T), T, 0, stream>>>(clsemb_w, attend, a_bf);
  gemm<0, 0>(a_bf, seq_repr, nullptr, scores, NC_, BS, 2 * H_, BS, (float)S_, stream);
  k_att_ctx<<<cdiv(NC_ * B_ * 2 * H_, T), T, 0, stream>>>(seq, scores, seq_repr, inp_bf);
  k_fill_inp<<<cdiv(NC_ * B_ * E_, T), T, 0, stream>>>(conv_seq, inp_bf);

  // ---- GRU decoder: input-side GEMM for all 6 steps at once ----
  gemm<0, 0>(inp_bf, dWih_bf, d_bih, gi_dec, NC_ * B_, 6 * H_, E_ + 2 * H_, 6 * H_, 1.f, stream);
  k_init_hdec<<<cdiv(B_ * 2 * H_, T), T, 0, stream>>>(seq_repr, h_dec);
  for (int c = 0; c < NC_; ++c) {
    gemm<0, 0>(h_dec, dWhh_bf, d_bhh, gh_dec, B_, 6 * H_, 2 * H_, 6 * H_, 1.f, stream);
    k_gru_gate<<<cdiv(B_ * 2 * H_, T), T, 0, stream>>>(c, gi_dec, gh_dec, h_dec);
    gemm<0, 0>(h_dec, projw_bf, proj_b, proj_out, B_, P_, 2 * H_, P_, 1.f, stream);
    k_cls_out<<<cdiv(B_, 32), 32, 0, stream>>>(c, proj_out, cls_w, cls_b, out);
  }
}
